// BaseAttention_58042188038650
// MI455X (gfx1250) — matrix operand order repaired and checked by reading, not compiled, not run on hardware
//
#include <hip/hip_runtime.h>

typedef __attribute__((ext_vector_type(16))) _Float16 v16h;
typedef __attribute__((ext_vector_type(8)))  float    v8f;
typedef unsigned int v4u __attribute__((ext_vector_type(4)));
typedef int          v8i __attribute__((ext_vector_type(8)));
typedef int          v4i __attribute__((ext_vector_type(4)));

#define SEQ 2048
#define DIM 64

// LDS layout (bytes) - double-buffered raw f32 tiles filled by TDM
#define RAWK0_OFF 0        // 32x64 f32 K tile, buffer 0 (8192 B)
#define RAWV0_OFF 8192     // 32x64 f32 V tile, buffer 0 (8192 B)
#define RAWK1_OFF 16384    // 32x64 f32 K tile, buffer 1 (8192 B)
#define RAWV1_OFF 24576    // 32x64 f32 V tile, buffer 1 (8192 B)
#define K16_OFF   32768    // 32x64 f16 K tile   (4096 B) row-major [key][d]
#define VT_OFF    36864    // 64x32 f16 V^T tile (4096 B) [d][key]
#define P_OFF     40960    // 4 waves x 16x32 f16 P tiles (4096 B)
#define SMEM_BYTES 45056

__device__ __forceinline__ v8f wmma_f16(v16h a, v16h b, v8f c) {
    // D = A(16x32 f16) * B(32x16 f16) + C(16x16 f32)
    return __builtin_amdgcn_wmma_f32_16x16x32_f16(
        false, a, false, b, (short)0, c, false, false);
}

// 16 consecutive floats (global or LDS) -> 16 halves, optional scale
__device__ __forceinline__ v16h load16_h(const float* __restrict__ p, float sc) {
    float4 x0 = ((const float4*)p)[0];
    float4 x1 = ((const float4*)p)[1];
    float4 x2 = ((const float4*)p)[2];
    float4 x3 = ((const float4*)p)[3];
    v16h a;
    a[0]=(_Float16)(x0.x*sc); a[1]=(_Float16)(x0.y*sc); a[2]=(_Float16)(x0.z*sc); a[3]=(_Float16)(x0.w*sc);
    a[4]=(_Float16)(x1.x*sc); a[5]=(_Float16)(x1.y*sc); a[6]=(_Float16)(x1.z*sc); a[7]=(_Float16)(x1.w*sc);
    a[8]=(_Float16)(x2.x*sc); a[9]=(_Float16)(x2.y*sc); a[10]=(_Float16)(x2.z*sc); a[11]=(_Float16)(x2.w*sc);
    a[12]=(_Float16)(x3.x*sc); a[13]=(_Float16)(x3.y*sc); a[14]=(_Float16)(x3.z*sc); a[15]=(_Float16)(x3.w*sc);
    return a;
}

// two groups of 8 consecutive floats -> A-fragment halves (elements 0-7 / 8-15)
__device__ __forceinline__ v16h load2x8_h(const float* __restrict__ p0,
                                          const float* __restrict__ p1, float sc) {
    float4 x0 = ((const float4*)p0)[0];
    float4 x1 = ((const float4*)p0)[1];
    float4 x2 = ((const float4*)p1)[0];
    float4 x3 = ((const float4*)p1)[1];
    v16h a;
    a[0]=(_Float16)(x0.x*sc); a[1]=(_Float16)(x0.y*sc); a[2]=(_Float16)(x0.z*sc); a[3]=(_Float16)(x0.w*sc);
    a[4]=(_Float16)(x1.x*sc); a[5]=(_Float16)(x1.y*sc); a[6]=(_Float16)(x1.z*sc); a[7]=(_Float16)(x1.w*sc);
    a[8]=(_Float16)(x2.x*sc); a[9]=(_Float16)(x2.y*sc); a[10]=(_Float16)(x2.z*sc); a[11]=(_Float16)(x2.w*sc);
    a[12]=(_Float16)(x3.x*sc); a[13]=(_Float16)(x3.y*sc); a[14]=(_Float16)(x3.z*sc); a[15]=(_Float16)(x3.w*sc);
    return a;
}

// Tensor Data Mover: 2D tile (tile1 rows x tile0 elems, 4-byte elements) -> LDS.
// D# bitfields per CDNA5 ISA; clang-23 6-arg builtin.
__device__ __forceinline__ void tdm_load_2d_f32(unsigned lds_addr, const void* gaddr,
                                                unsigned dim0, unsigned dim1,
                                                unsigned tile0, unsigned tile1,
                                                unsigned stride0) {
    unsigned long long ga = (unsigned long long)(uintptr_t)gaddr;
    v4u g0;
    g0[0] = 1u;                                            // count=1 (valid user D#)
    g0[1] = lds_addr;                                      // lds_addr [63:32]
    g0[2] = (unsigned)(ga & 0xffffffffu);                  // global_addr lo
    g0[3] = (unsigned)((ga >> 32) & 0x01ffffffu)           // global_addr [56:32]
          | (2u << 30);                                    // type = 2 ("image")
    v8i g1;
    g1[0] = (int)(2u << 16);                               // wg_mask=0, data_size=2 (4B)
    g1[1] = (int)((dim0 & 0xffffu) << 16);                 // tensor_dim0 [15:0]
    g1[2] = (int)(((dim0 >> 16) & 0xffffu) | ((dim1 & 0xffffu) << 16));
    g1[3] = (int)(((dim1 >> 16) & 0xffffu) | ((tile0 & 0xffffu) << 16));  // tile_dim0
    g1[4] = (int)(tile1 & 0xffffu);                        // tile_dim1, tile_dim2=0
    g1[5] = (int)stride0;                                  // tensor_dim0_stride lo32
    g1[6] = 0;                                             // stride0 hi16 | stride1 lo16
    g1[7] = 0;                                             // stride1 hi32
    v4i z4 = {};
    v8i z8 = {};
    __builtin_amdgcn_tensor_load_to_lds(g0, g1, z4, z4, z8, 0);
}

// One 32-key tile of flash attention for one wave.  Scores arrive already in the
// log2 domain (Q pre-scaled by log2e/sqrt(D)).  MASKED instantiated only for the
// 1-2 diagonal tiles; interior tiles skip all v_cmp/v_cndmask work.
// Row sums are accumulated on the XDL via a ones-column WMMA (lsum) instead of
// ds_bpermute butterflies.
template<bool MASKED>
__device__ __forceinline__ void attn_tile(int kt, int qt, int row_, int hlf,
                                          v16h aq0, v16h aq1,
                                          const _Float16* __restrict__ k16,
                                          const _Float16* __restrict__ vT,
                                          _Float16* __restrict__ myP,
                                          int swz0, int swz1,
                                          float* __restrict__ m,
                                          v8f* __restrict__ acc, v8f& lsum)
{
    // ---- scores S = Q*K^T : two 16x16 C tiles from contiguous ds_load_b128 ----
    const _Float16* kr0 = k16 + (size_t)(row_) * DIM;        // key sub-tile 0
    const _Float16* kr1 = k16 + (size_t)(16 + row_) * DIM;   // key sub-tile 1
    v16h bk00 = *(const v16h*)(kr0 +  0 + hlf * 16);
    v16h bk01 = *(const v16h*)(kr0 + 32 + hlf * 16);
    v16h bk10 = *(const v16h*)(kr1 +  0 + hlf * 16);
    v16h bk11 = *(const v16h*)(kr1 + 32 + hlf * 16);
    v8f z = {};
    v8f s0 = wmma_f16(aq0, bk00, z);
    s0     = wmma_f16(aq1, bk01, s0);
    v8f s1 = wmma_f16(aq0, bk10, z);
    s1     = wmma_f16(aq1, bk11, s1);

    // ---- (optional) causal mask + online softmax in log2 domain ----
    #pragma unroll
    for (int r = 0; r < 8; ++r) {
        float e0 = s0[r], e1 = s1[r];
        if (MASKED) {
            const int qg = qt + r + hlf * 8;
            if (kt + row_      > qg) e0 = -3.0e38f;
            if (kt + 16 + row_ > qg) e1 = -3.0e38f;
        }
        float mx = fmaxf(e0, e1);
        mx = fmaxf(mx, __shfl_xor(mx, 1, 32));
        mx = fmaxf(mx, __shfl_xor(mx, 2, 32));
        mx = fmaxf(mx, __shfl_xor(mx, 4, 32));
        mx = fmaxf(mx, __shfl_xor(mx, 8, 32));

        const float mn   = fmaxf(m[r], mx);
        const float corr = exp2f(m[r] - mn);
        m[r] = mn;

        const float p0 = exp2f(e0 - mn);
        const float p1 = exp2f(e1 - mn);

        myP[(r + hlf * 8) * 32 + swz0] = (_Float16)p0;
        myP[(r + hlf * 8) * 32 + swz1] = (_Float16)p1;

        // rescale all running accumulators (output columns + row-sum column)
        lsum[r] *= corr;
        #pragma unroll
        for (int c = 0; c < 4; ++c) acc[c][r] *= corr;
    }

    // per-wave P region; DS ops are in-order per wave, just drain the counter
    asm volatile("s_wait_dscnt 0x0" ::: "memory");

    // reload P as A-fragment: one contiguous 32B run per lane (contiguous layout)
    v16h ap = *(const v16h*)(myP + row_ * 32 + hlf * 16);

    // ---- O += P * V : B fragments are contiguous runs of V^T ----
    v16h bv0 = *(const v16h*)(vT + (0 * 16 + row_) * 32 + hlf * 16);
    v16h bv1 = *(const v16h*)(vT + (1 * 16 + row_) * 32 + hlf * 16);
    v16h bv2 = *(const v16h*)(vT + (2 * 16 + row_) * 32 + hlf * 16);
    v16h bv3 = *(const v16h*)(vT + (3 * 16 + row_) * 32 + hlf * 16);
    acc[0] = wmma_f16(ap, bv0, acc[0]);
    acc[1] = wmma_f16(ap, bv1, acc[1]);
    acc[2] = wmma_f16(ap, bv2, acc[2]);
    acc[3] = wmma_f16(ap, bv3, acc[3]);

    // row-sum on the XDL: lsum += P * ones  (every column holds sum_k p[r][k])
    v16h ones;
    #pragma unroll
    for (int i = 0; i < 16; ++i) ones[i] = (_Float16)1.0f;
    lsum = wmma_f16(ap, ones, lsum);
}

__global__ __launch_bounds__(128)
void flash_attn_tdm_gfx1250(const float* __restrict__ Q, const float* __restrict__ K,
                            const float* __restrict__ V, float* __restrict__ O)
{
    __shared__ __align__(16) unsigned char smem[SMEM_BYTES];
    _Float16* k16  = (_Float16*)(smem + K16_OFF);
    _Float16* vT   = (_Float16*)(smem + VT_OFF);
    _Float16* ldsP = (_Float16*)(smem + P_OFF);

    const int lane = threadIdx.x & 31;
    const int wave = threadIdx.x >> 5;
    const int row_ = lane & 15;       // N (cols) for B/C frags, M for A frags
    const int hlf  = lane >> 4;       // 0: lanes 0-15, 1: lanes 16-31

    const int bh  = blockIdx.y;                 // fused (b*H + h)
    const int qtb = blockIdx.x * 64;            // block's 64-row q tile
    const int qt  = qtb + wave * 16;            // this wave's 16-row q tile

    const size_t headOff = (size_t)bh * SEQ * DIM;
    const float* qh = Q + headOff;
    const float* kh = K + headOff;
    const float* vh = V + headOff;
    float*       oh = O + headOff;

    _Float16* myP = ldsP + wave * 16 * 32;
    // flat->LDS: low 32 bits of a flat LDS address are the LDS byte address
    const unsigned ldsBase = (unsigned)(uintptr_t)(void*)smem;

    // ---- Q tile -> two A-fragments; log2e/sqrt(D) folded in (log2-domain scores) ----
    const float* qrow = qh + (size_t)(qt + row_) * DIM;
    const float  qscale = 0.125f * 1.44269504088896340736f;
    v16h aq0 = load2x8_h(qrow +  0 + hlf*16, qrow +  8 + hlf*16, qscale);
    v16h aq1 = load2x8_h(qrow + 32 + hlf*16, qrow + 40 + hlf*16, qscale);

    // P layout: swz(k) = k (contiguous)
    const int swz0 = row_;                             // swz(row_)
    const int swz1 = row_ | 16;                        // swz(16+row_)

    float m[8];
    v8f acc[4], lsum;
    #pragma unroll
    for (int r = 0; r < 8; ++r) m[r] = -3.0e38f;
    {
        v8f z = {};
        #pragma unroll
        for (int c = 0; c < 4; ++c) acc[c] = z;
        lsum = z;
    }

    const int jend = (qtb + 63) >> 5;   // block-uniform trip count (barrier-legal)

    // prologue: kick off TDM for tile 0 into buffer 0
    if (wave == 0) {
        tdm_load_2d_f32(ldsBase + RAWK0_OFF, kh, DIM, SEQ, DIM, 32, DIM);
        tdm_load_2d_f32(ldsBase + RAWV0_OFF, vh, DIM, SEQ, DIM, 32, DIM);
    }

    for (int j = 0; j <= jend; ++j) {
        const int kt  = j * 32;
        const int buf = j & 1;
        const float* rawK = (const float*)(smem + (buf ? RAWK1_OFF : RAWK0_OFF));
        const float* rawV = (const float*)(smem + (buf ? RAWV1_OFF : RAWV0_OFF));

        if (wave == 0) __builtin_amdgcn_s_wait_tensorcnt(0);  // tile j DMA done
        __syncthreads();  // raw f32 tile j visible; compute of j-1 fully retired

        // overlap: launch tile j+1 DMA into the other buffer during convert+compute
        if (wave == 0 && j < jend) {
            const int kt2 = kt + 32;
            const unsigned nk = buf ? RAWK0_OFF : RAWK1_OFF;
            const unsigned nv = buf ? RAWV0_OFF : RAWV1_OFF;
            tdm_load_2d_f32(ldsBase + nk, kh + (size_t)kt2 * DIM,
                            DIM, (unsigned)(SEQ - kt2), DIM, 32, DIM);
            tdm_load_2d_f32(ldsBase + nv, vh + (size_t)kt2 * DIM,
                            DIM, (unsigned)(SEQ - kt2), DIM, 32, DIM);
        }

        // ---- cooperative f32->f16 convert: 128 threads x 16 elems ----
        {
            const int t   = threadIdx.x;
            const int key = t >> 2;            // 0..31
            const int dq  = (t & 3) << 4;      // 0,16,32,48
            v16h hk = load16_h(rawK + key * DIM + dq, 1.0f);
            *(v16h*)(k16 + key * DIM + dq) = hk;               // [key][d] f16
            v16h hv = load16_h(rawV + key * DIM + dq, 1.0f);
            #pragma unroll
            for (int i = 0; i < 16; ++i)
                vT[(dq + i) * 32 + key] = hv[i];               // [d][key] f16
        }
        __syncthreads();   // f16 tiles ready

        // wave-uniform tile classification vs this wave's q range [qt, qt+15]
        if (kt <= qt + 15) {
            if (kt + 31 <= qt)
                attn_tile<false>(kt, qt, row_, hlf, aq0, aq1, k16, vT, myP,
                                 swz0, swz1, m, acc, lsum);  // fully causal-interior
            else
                attn_tile<true >(kt, qt, row_, hlf, aq0, aq1, k16, vT, myP,
                                 swz0, swz1, m, acc, lsum);  // diagonal boundary tile
        }
        // else: tile entirely above the diagonal for this wave -> skip compute
    }

    // ---- normalize and store (lsum[r] = row sum, identical in every column) ----
    float invl[8];
    #pragma unroll
    for (int r = 0; r < 8; ++r) invl[r] = 1.0f / lsum[r];

    #pragma unroll
    for (int c = 0; c < 4; ++c) {
        #pragma unroll
        for (int r = 0; r < 8; ++r) {
            oh[(size_t)(qt + r + hlf * 8) * DIM + c * 16 + row_] = acc[c][r] * invl[r];
        }
    }
}

extern "C" void kernel_launch(void* const* d_in, const int* in_sizes, int n_in,
                              void* d_out, int out_size, void* d_ws, size_t ws_size,
                              hipStream_t stream) {
    const float* q = (const float*)d_in[0];
    const float* k = (const float*)d_in[1];
    const float* v = (const float*)d_in[2];
    // d_in[3] is the causal mask; computed analytically in-kernel.
    float* o = (float*)d_out;

    const int B = 2, H = 16;
    dim3 grid(SEQ / 64, B * H);   // (32, 32)
    dim3 block(128);              // 4 wave32 waves, one 16-row q tile each
    flash_attn_tdm_gfx1250<<<grid, block, 0, stream>>>(q, k, v, o);
}
